// head3_56813827391728
// MI455X (gfx1250) — matrix operand order repaired and checked by reading, not compiled, not run on hardware
//
#include <hip/hip_runtime.h>
#include <cstddef>

// ---------- types ----------
typedef _Float16 v16h __attribute__((ext_vector_type(16)));
typedef float    v8f  __attribute__((ext_vector_type(8)));
typedef _Float16 h2   __attribute__((ext_vector_type(2)));

#define NPIX      32768            // B*H*W = 32*32*32
#define INNER     324              // 18*18
#define ACT_ELEMS (NPIX * INNER)   // floats per activation buffer

// ============================================================
// WMMA wrappers
// ============================================================
__device__ inline v8f wmma_f32c(v16h a, v16h b, v8f c) {
#if defined(__HIP_DEVICE_COMPILE__)
  return __builtin_amdgcn_wmma_f32_16x16x32_f16(false, a, false, b, (short)0, c,
                                                false, false);
#else
  (void)a; (void)b; return c;
#endif
}

#if defined(__HIP_DEVICE_COMPILE__) && \
    __has_builtin(__builtin_amdgcn_wmma_f16_16x16x32_f16)
#define HAVE_WMMA_F16C 1
typedef _Float16 v8h __attribute__((ext_vector_type(8)));
__device__ inline v8h wmma_f16c(v16h a, v16h b, v8h c) {
  return __builtin_amdgcn_wmma_f16_16x16x32_f16(false, a, false, b, (short)0, c,
                                                false, false);
}
#endif

// A fragment (16x32, f16) from row-major [32][32] f16 storage.
// Lane L holds row M=L&15; K = (p>=4)*16 + (L>=16)*8 + (p&3)*2 + pos
__device__ inline v16h load_frag_A(const _Float16* m, int lane) {
  const int row = lane & 15;
  const int hi  = (lane >> 4) & 1;
  v16h a;
#pragma unroll
  for (int p = 0; p < 8; ++p) {
    const int k = ((p >> 2) << 4) + (hi << 3) + ((p & 3) << 1);
    h2 v = *(const h2*)(m + row * 32 + k);
    a[2 * p]     = v.x;
    a[2 * p + 1] = v.y;
  }
  return a;
}

// B fragment (32x16, f16) from TRANSPOSED [n][k] f16 storage (stride 32).
// Lane L holds col N=L&15; K = (p>=4)*16 + (L>=16)*8 + (p&3)*2 + pos
__device__ inline v16h load_frag_B(const _Float16* mt, int lane) {
  const int n  = lane & 15;
  const int hi = (lane >> 4) & 1;
  v16h b;
#pragma unroll
  for (int p = 0; p < 8; ++p) {
    const int k = ((p >> 2) << 4) + (hi << 3) + ((p & 3) << 1);
    h2 v = *(const h2*)(mt + n * 32 + k);
    b[2 * p]     = v.x;
    b[2 * p + 1] = v.y;
  }
  return b;
}

// Store 16x16 f32 C fragment (tile mt,nt) into row-major [32][32] f16 LDS.
// f32 C layout: VGPR i -> row = i + 8*(lane>=16), col = lane&15
__device__ inline void store_Cf32(const v8f& c, _Float16* ldsZ, int mt, int nt,
                                  int lane) {
  const int col = nt * 16 + (lane & 15);
  const int rb  = mt * 16 + (((lane >> 4) & 1) << 3);
#pragma unroll
  for (int i = 0; i < 8; ++i) ldsZ[(rb + i) * 32 + col] = (_Float16)c[i];
}

#ifdef HAVE_WMMA_F16C
// f16 C/D layout: VGPR i -> rows M=2i (lo),2i+1 (hi); +8 for lanes>=16
// element j of v8h -> row = base + j
__device__ inline void store_Cf16(const v8h& c, _Float16* ldsZ, int mt, int nt,
                                  int lane) {
  const int col = nt * 16 + (lane & 15);
  const int rb  = mt * 16 + (((lane >> 4) & 1) << 3);
#pragma unroll
  for (int j = 0; j < 8; ++j) ldsZ[(rb + j) * 32 + col] = c[j];
}
#endif

// One bilinear tap: acc(32x32) += Upad(32x32) * Xpad(32x32 in ldsX^T) * Vpad
__device__ inline void bilinear_step(const _Float16* U, const _Float16* Vt,
                                     const _Float16* ldsX, _Float16* ldsZ,
                                     int lane, v8f& a00, v8f& a01, v8f& a10,
                                     v8f& a11) {
  v16h aU0 = load_frag_A(U, lane);
  v16h aU1 = load_frag_A(U + 512, lane);
  v16h bX0 = load_frag_B(ldsX, lane);
  v16h bX1 = load_frag_B(ldsX + 512, lane);
#ifdef HAVE_WMMA_F16C
  v8h hz = {0, 0, 0, 0, 0, 0, 0, 0};
  v8h z00 = wmma_f16c(aU0, bX0, hz);
  v8h z01 = wmma_f16c(aU0, bX1, hz);
  v8h z10 = wmma_f16c(aU1, bX0, hz);
  v8h z11 = wmma_f16c(aU1, bX1, hz);
  store_Cf16(z00, ldsZ, 0, 0, lane);
  store_Cf16(z01, ldsZ, 0, 1, lane);
  store_Cf16(z10, ldsZ, 1, 0, lane);
  store_Cf16(z11, ldsZ, 1, 1, lane);
#else
  v8f zz = {0, 0, 0, 0, 0, 0, 0, 0};
  v8f z00 = wmma_f32c(aU0, bX0, zz);
  v8f z01 = wmma_f32c(aU0, bX1, zz);
  v8f z10 = wmma_f32c(aU1, bX0, zz);
  v8f z11 = wmma_f32c(aU1, bX1, zz);
  store_Cf32(z00, ldsZ, 0, 0, lane);
  store_Cf32(z01, ldsZ, 0, 1, lane);
  store_Cf32(z10, ldsZ, 1, 0, lane);
  store_Cf32(z11, ldsZ, 1, 1, lane);
#endif
  __syncthreads();  // single-wave WG: barrier -> S_NOP, keeps ds ordering
  v16h aZ0 = load_frag_A(ldsZ, lane);
  v16h aZ1 = load_frag_A(ldsZ + 512, lane);
  v16h bV0 = load_frag_B(Vt, lane);
  v16h bV1 = load_frag_B(Vt + 512, lane);
  a00 = wmma_f32c(aZ0, bV0, a00);
  a01 = wmma_f32c(aZ0, bV1, a01);
  a10 = wmma_f32c(aZ1, bV0, a10);
  a11 = wmma_f32c(aZ1, bV1, a11);
}

// Store the 18x18 valid region of the 32x32 accumulator to global f32.
__device__ inline void store_Y(const v8f& a00, const v8f& a01, const v8f& a10,
                               const v8f& a11, float* dst, int lane) {
  const int col = lane & 15;
  const int rb  = ((lane >> 4) & 1) << 3;
#pragma unroll
  for (int i = 0; i < 8; ++i) {
    const int r = rb + i;
    if (col < 18) dst[r * 18 + col] = a00[i];
    if (col < 2)  dst[r * 18 + 16 + col] = a01[i];
    const int r2 = 16 + r;
    if (r2 < 18) {
      if (col < 18) dst[r2 * 18 + col] = a10[i];
      if (col < 2)  dst[r2 * 18 + 16 + col] = a11[i];
    }
  }
}

// ============================================================
// Generic tensor-conv: 18x18 activations, 1 or 9 taps.
// ONE WAVE per pixel, single-wave workgroup (barriers ~free).
// ============================================================
__global__ __launch_bounds__(32) void conv_bilinear_kernel(
    const float* __restrict__ in, const _Float16* __restrict__ Uf,
    const _Float16* __restrict__ Vt, float* __restrict__ out, int taps) {
  __shared__ _Float16 lds[2][1024];
  const int lane = threadIdx.x & 31;
  const int pix  = blockIdx.x;
  const int b = pix >> 10, h = (pix >> 5) & 31, w = pix & 31;
  _Float16* ldsX = &lds[0][0];
  _Float16* ldsZ = &lds[1][0];
  v8f a00 = {0, 0, 0, 0, 0, 0, 0, 0}, a01 = a00, a10 = a00, a11 = a00;

  for (int t = 0; t < taps; ++t) {
    const int di = (taps == 9) ? (t / 3 - 1) : 0;
    const int dj = (taps == 9) ? (t % 3 - 1) : 0;
    const int hh = h + di, ww = w + dj;
    const bool ok = ((unsigned)hh < 32u) && ((unsigned)ww < 32u);
    const float* src = in + (((size_t)b << 10) + (hh << 5) + ww) * INNER;
    // prefetch next tap's activation tile (global_prefetch_b8, speculative)
    if (taps == 9 && t + 1 < 9) {
      const int hh2 = h + (t + 1) / 3 - 1, ww2 = w + (t + 1) % 3 - 1;
      if (((unsigned)hh2 < 32u) && ((unsigned)ww2 < 32u)) {
        const float* nsrc = in + (((size_t)b << 10) + (hh2 << 5) + ww2) * INNER;
        __builtin_prefetch(nsrc + lane * 16, 0, 1);
      }
    }
    __syncthreads();
    {  // stage X^T into LDS: lane owns column col=lane (zero-padded to 32x32)
      const int col = lane;
#pragma unroll
      for (int r = 0; r < 32; r += 2) {
        float v0 = 0.f, v1 = 0.f;
        if (ok && col < 18 && r < 18) {
          v0 = src[r * 18 + col];
          if (r + 1 < 18) v1 = src[(r + 1) * 18 + col];
        }
        h2 pk; pk.x = (_Float16)v0; pk.y = (_Float16)v1;
        *(h2*)(ldsX + col * 32 + r) = pk;
      }
    }
    __syncthreads();
    bilinear_step(Uf + t * 1024, Vt + t * 1024, ldsX, ldsZ, lane, a00, a01,
                  a10, a11);
  }
  store_Y(a00, a01, a10, a11, out + (size_t)pix * INNER, lane);
}

// ============================================================
// conv_f0: input (B,8,16,16,32,32), 9 taps x 8 channels, 16x16 -> 18x18
// ============================================================
__global__ __launch_bounds__(32) void conv_f0_kernel(
    const float* __restrict__ x, const _Float16* __restrict__ Uf,
    const _Float16* __restrict__ Vt, float* __restrict__ out) {
  __shared__ _Float16 lds[2][1024];
  const int lane = threadIdx.x & 31;
  const int pix  = blockIdx.x;
  const int b = pix >> 10, h = (pix >> 5) & 31, w = pix & 31;
  _Float16* ldsX = &lds[0][0];
  _Float16* ldsZ = &lds[1][0];
  v8f a00 = {0, 0, 0, 0, 0, 0, 0, 0}, a01 = a00, a10 = a00, a11 = a00;

  for (int t = 0; t < 9; ++t) {
    const int hh = h + t / 3 - 1, ww = w + t % 3 - 1;
    const bool ok = ((unsigned)hh < 32u) && ((unsigned)ww < 32u);
    for (int ch = 0; ch < 8; ++ch) {
      __syncthreads();
      {  // stage X^T (16x16, zero-padded): element (p,q) -> ldsX[q*32+p]
        const int col = lane;  // q
        const float* base = x + ((size_t)(b * 8 + ch) << 8) * 1024 +
                            ((size_t)hh << 5) + ww;
#pragma unroll
        for (int p = 0; p < 32; p += 2) {
          float v0 = 0.f, v1 = 0.f;
          if (ok && col < 16 && p < 16) {
            v0 = base[(size_t)(p * 16 + col) << 10];
            v1 = base[(size_t)((p + 1) * 16 + col) << 10];
          }
          h2 pk; pk.x = (_Float16)v0; pk.y = (_Float16)v1;
          *(h2*)(ldsX + col * 32 + p) = pk;
        }
      }
      __syncthreads();
      const int m = ch * 9 + t;
      bilinear_step(Uf + m * 1024, Vt + m * 1024, ldsX, ldsZ, lane, a00, a01,
                    a10, a11);
    }
  }
  store_Y(a00, a01, a10, a11, out + (size_t)pix * INNER, lane);
}

// ============================================================
// Weight prep: f32 -> padded f16, U row-major, V transposed.
// Layout (32x32 f16 mats): [0,72) U0 | [72,144) V0t | [144,153) U1 |
//                          [153,162) V1t | [162,189) U3 | [189,216) V3t
// ============================================================
__global__ void prep_weights_kernel(const float* __restrict__ U0,
                                    const float* __restrict__ V0,
                                    const float* __restrict__ U1,
                                    const float* __restrict__ V1,
                                    const float* __restrict__ U3,
                                    const float* __restrict__ V3,
                                    _Float16* __restrict__ dst) {
  const int idx = blockIdx.x * blockDim.x + threadIdx.x;
  if (idx >= 216 * 1024) return;
  const int mat = idx >> 10;
  const int i   = idx & 1023;
  const int a   = i >> 5;   // dst row
  const int c   = i & 31;   // dst col
  float val = 0.f;
  if (mat < 72) {                     // U0: 18x16 row-major
    if (a < 18 && c < 16) val = U0[mat * 288 + a * 16 + c];
  } else if (mat < 144) {             // V0t: V0 is 16x18 (k,n); dst[n][k]
    const int m = mat - 72;
    if (c < 16 && a < 18) val = V0[m * 288 + c * 18 + a];
  } else if (mat < 153) {             // U1: 18x18
    const int m = mat - 144;
    if (a < 18 && c < 18) val = U1[m * 324 + a * 18 + c];
  } else if (mat < 162) {             // V1t
    const int m = mat - 153;
    if (c < 18 && a < 18) val = V1[m * 324 + c * 18 + a];
  } else if (mat < 189) {             // U3: 18x18
    const int m = mat - 162;
    if (a < 18 && c < 18) val = U3[m * 324 + a * 18 + c];
  } else {                            // V3t
    const int m = mat - 189;
    if (c < 18 && a < 18) val = V3[m * 324 + c * 18 + a];
  }
  dst[idx] = (_Float16)val;
}

// ============================================================
// BatchNorm helpers
// ============================================================
__global__ void zero_f32_kernel(float* __restrict__ p, int n) {
  const int i = blockIdx.x * blockDim.x + threadIdx.x;
  if (i < n) p[i] = 0.f;
}

// 256 blocks x 128 threads; block handles 128 pixels; thread t covers
// positions t, t+128, t+256 (coalesced over the inner 324 positions)
__global__ void bn_stats_kernel(const float* __restrict__ act,
                                float* __restrict__ stats) {
  const int t  = threadIdx.x;
  const int p0 = blockIdx.x * 128;
  float s0 = 0, s1 = 0, s2 = 0, q0 = 0, q1 = 0, q2 = 0;
  for (int p = p0; p < p0 + 128; ++p) {
    const float* row = act + (size_t)p * INNER;
    float v0 = row[t];       s0 += v0; q0 += v0 * v0;
    float v1 = row[t + 128]; s1 += v1; q1 += v1 * v1;
    if (t < 68) { float v2 = row[t + 256]; s2 += v2; q2 += v2 * v2; }
  }
  atomicAdd(&stats[t], s0);               atomicAdd(&stats[INNER + t], q0);
  atomicAdd(&stats[t + 128], s1);         atomicAdd(&stats[INNER + t + 128], q1);
  if (t < 68) { atomicAdd(&stats[t + 256], s2);
                atomicAdd(&stats[INNER + t + 256], q2); }
}

// normalize (+ optional PReLU, + optional residual) in-place
__global__ void bn_norm_kernel(float* __restrict__ io,
                               const float* __restrict__ stats,
                               const float* __restrict__ gamma,
                               const float* __restrict__ beta,
                               const float* __restrict__ alphap,
                               const float* __restrict__ res, int n) {
  const int idx = blockIdx.x * blockDim.x + threadIdx.x;
  if (idx >= n) return;
  const int pos = idx % INNER;
  const float m = stats[pos] * (1.f / 32768.f);
  const float v = stats[INNER + pos] * (1.f / 32768.f) - m * m;
  float y = gamma[pos] * (io[idx] - m) * rsqrtf(v + 1e-5f) + beta[pos];
  if (alphap) { const float a = *alphap; y = (y >= 0.f) ? y : a * y; }
  if (res) y += res[idx];
  io[idx] = y;
}

// fcn head: out[b,o,h,w] = sum_{pq} act[pix,p,q]*Wf[o,p,q] + bf[o]
__global__ void head_kernel(const float* __restrict__ act,
                            const float* __restrict__ Wf,
                            const float* __restrict__ bf,
                            float* __restrict__ out) {
  const int idx = blockIdx.x * blockDim.x + threadIdx.x;
  if (idx >= 32 * 10 * 1024) return;
  const int hw = idx & 1023;
  const int o  = (idx >> 10) % 10;
  const int b  = idx / 10240;
  const float* a = act + (((size_t)b << 10) + hw) * INNER;
  const float* wr = Wf + o * INNER;
  float s = bf[o];
  for (int k = 0; k < INNER; ++k) s += a[k] * wr[k];
  out[idx] = s;
}

// ============================================================
// launcher
// ============================================================
extern "C" void kernel_launch(void* const* d_in, const int* in_sizes, int n_in,
                              void* d_out, int out_size, void* d_ws,
                              size_t ws_size, hipStream_t stream) {
  const float* x     = (const float*)d_in[0];
  const float* U0    = (const float*)d_in[1];
  const float* V0    = (const float*)d_in[2];
  // d_in[3] = b0 : cancelled exactly by BN mean subtraction
  const float* U1    = (const float*)d_in[4];
  const float* V1    = (const float*)d_in[5];
  // d_in[6] = b1 : cancelled by BN
  const float* U3    = (const float*)d_in[7];
  const float* V3    = (const float*)d_in[8];
  // d_in[9] = b3 : cancelled by BN
  const float* Wf    = (const float*)d_in[10];
  const float* bf    = (const float*)d_in[11];
  const float* gamma = (const float*)d_in[12];
  const float* beta  = (const float*)d_in[13];
  const float* alpha = (const float*)d_in[14];
  float* out = (float*)d_out;

  _Float16* wb = (_Float16*)d_ws;
  float* stats = (float*)((char*)d_ws + 216 * 1024 * sizeof(_Float16));
  float* bufH  = stats + 2 * INNER;
  float* bufID = bufH + ACT_ELEMS;
  float* bufT  = bufID + ACT_ELEMS;
  float* bufT2 = bufT + ACT_ELEMS;

  const _Float16* wU0 = wb;
  const _Float16* wV0 = wb + 72 * 1024;
  const _Float16* wU1 = wb + 144 * 1024;
  const _Float16* wV1 = wb + 153 * 1024;
  const _Float16* wU3 = wb + 162 * 1024;
  const _Float16* wV3 = wb + 189 * 1024;

  prep_weights_kernel<<<864, 256, 0, stream>>>(U0, V0, U1, V1, U3, V3, wb);

  auto bn_all = [&](float* buf, int ibn, int ia, const float* res) {
    zero_f32_kernel<<<3, 256, 0, stream>>>(stats, 2 * INNER);
    bn_stats_kernel<<<256, 128, 0, stream>>>(buf, stats);
    bn_norm_kernel<<<41472, 256, 0, stream>>>(
        buf, stats, gamma + ibn * INNER, beta + ibn * INNER,
        (ia >= 0) ? (alpha + ia) : nullptr, res, (int)ACT_ELEMS);
  };

  // f0
  conv_f0_kernel<<<NPIX, 32, 0, stream>>>(x, wU0, wV0, bufH);
  bn_all(bufH, 0, 0, nullptr);
  // s1 identity
  conv_bilinear_kernel<<<NPIX, 32, 0, stream>>>(bufH, wU1, wV1, bufID, 1);
  bn_all(bufID, 1, -1, nullptr);
  // resblock 1
  conv_bilinear_kernel<<<NPIX, 32, 0, stream>>>(bufH, wU1 + 1024, wV1 + 1024, bufT, 1);
  bn_all(bufT, 2, 1, nullptr);
  conv_bilinear_kernel<<<NPIX, 32, 0, stream>>>(bufT, wU3, wV3, bufT2, 9);
  bn_all(bufT2, 3, 2, nullptr);
  conv_bilinear_kernel<<<NPIX, 32, 0, stream>>>(bufT2, wU1 + 2 * 1024, wV1 + 2 * 1024, bufT, 1);
  bn_all(bufT, 4, -1, bufID);  // x11 in bufT
  // s2 identity
  conv_bilinear_kernel<<<NPIX, 32, 0, stream>>>(bufT, wU1 + 3 * 1024, wV1 + 3 * 1024, bufID, 1);
  bn_all(bufID, 5, -1, nullptr);  // ID2 in bufID
  // resblock 2
  conv_bilinear_kernel<<<NPIX, 32, 0, stream>>>(bufT, wU1 + 4 * 1024, wV1 + 4 * 1024, bufT2, 1);
  bn_all(bufT2, 6, 3, nullptr);
  conv_bilinear_kernel<<<NPIX, 32, 0, stream>>>(bufT2, wU3 + 9 * 1024, wV3 + 9 * 1024, bufT, 9);
  bn_all(bufT, 7, 4, nullptr);
  conv_bilinear_kernel<<<NPIX, 32, 0, stream>>>(bufT, wU1 + 5 * 1024, wV1 + 5 * 1024, bufT2, 1);
  bn_all(bufT2, 8, -1, bufID);  // x21 in bufT2 (ID2 preserved; s3 unused -> skipped)
  // resblock 3 (bug-faithful: residual = ID2 again)
  conv_bilinear_kernel<<<NPIX, 32, 0, stream>>>(bufT2, wU1 + 7 * 1024, wV1 + 7 * 1024, bufT, 1);
  bn_all(bufT, 10, 5, nullptr);
  conv_bilinear_kernel<<<NPIX, 32, 0, stream>>>(bufT, wU3 + 18 * 1024, wV3 + 18 * 1024, bufT2, 9);
  bn_all(bufT2, 11, 6, nullptr);
  conv_bilinear_kernel<<<NPIX, 32, 0, stream>>>(bufT2, wU1 + 8 * 1024, wV1 + 8 * 1024, bufT, 1);
  bn_all(bufT, 12, -1, bufID);  // x31 in bufT
  // head
  head_kernel<<<1280, 256, 0, stream>>>(bufT, Wf, bf, out);

  (void)in_sizes; (void)n_in; (void)out_size; (void)ws_size;
}
